// SimpleGNN_3229815407289
// MI455X (gfx1250) — matrix operand order repaired and checked by reading, not compiled, not run on hardware
//
#include <hip/hip_runtime.h>
#include <cstdint>
#include <cstddef>

// Problem constants from the reference
#define BATCH 4
#define NNODE 10000
#define DIN   128
#define HDIM  256
#define MROWS (BATCH * NNODE)   // 40000 rows, 2500 M-tiles of 16

typedef __attribute__((ext_vector_type(16))) __bf16       v16bf;
typedef __attribute__((ext_vector_type(8)))  float        v8f;
typedef __attribute__((ext_vector_type(8)))  unsigned int v8u;
typedef __attribute__((ext_vector_type(4)))  unsigned int v4u;

__device__ __forceinline__ unsigned int f2bf(float f) {
  // round-to-nearest-even f32 -> bf16
  unsigned int u = __float_as_uint(f);
  return (u + 0x7FFFu + ((u >> 16) & 1u)) >> 16;
}

// ---------------------------------------------------------------------------
// Pack W [K x 256] f32 row-major into WMMA B-fragment order (bf16 pairs):
//   dword index = ((tn*(K/32) + kt)*32 + lane)*8 + v
//   lane<16: N=tn*16+lane,      K = kt*32 + 16*(v>>2) + 2*(v&3), +1
//   lane>=16: N=tn*16+lane-16,  K = kt*32 + 8 + 16*(v>>2) + 2*(v&3), +1
// ---------------------------------------------------------------------------
__global__ void pack_w_kernel(const float* __restrict__ W,
                              unsigned int* __restrict__ Wp, int K) {
  const int ksteps = K >> 5;
  const int total  = 16 * ksteps * 32 * 8;
  int idx = blockIdx.x * blockDim.x + threadIdx.x;
  if (idx >= total) return;
  const int v  = idx & 7;
  const int l  = (idx >> 3) & 31;
  const int kt = (idx >> 8) % ksteps;
  const int tn = idx / (256 * ksteps);
  const int n  = tn * 16 + (l & 15);
  const int kb = kt * 32 + ((l & 16) ? 8 : 0) + ((v & 4) ? 16 : 0) + 2 * (v & 3);
  const unsigned int lo = f2bf(W[(size_t)kb * HDIM + n]);
  const unsigned int hi = f2bf(W[(size_t)(kb + 1) * HDIM + n]);
  Wp[idx] = lo | (hi << 16);
}

// fp32 -> bf16 bulk convert (activations)
__global__ void f32_to_bf16_kernel(const float* __restrict__ in,
                                   unsigned short* __restrict__ out, int n) {
  int i = blockIdx.x * blockDim.x + threadIdx.x;
  if (i < n) out[i] = (unsigned short)f2bf(in[i]);
}

// grid-stride zero fill
__global__ void zero_f32_kernel(float* __restrict__ p, size_t n) {
  size_t i = (size_t)blockIdx.x * blockDim.x + threadIdx.x;
  size_t stride = (size_t)gridDim.x * blockDim.x;
  for (; i < n; i += stride) p[i] = 0.0f;
}

// ---------------------------------------------------------------------------
// C[M,256] = relu(A_bf16[M,K] @ W_packed + bias)
// One wave computes a 16x64 strip: 4 accumulators share one A fragment,
// so each K-step is 2 A loads + 8 B loads feeding 4 v_wmma (2.5 b128 / wmma).
// ---------------------------------------------------------------------------
__global__ __launch_bounds__(256) void gemm_bias_relu_bf16(
    const unsigned short* __restrict__ Abf,   // [MROWS, K] bf16
    const unsigned int*   __restrict__ Wp,    // packed B fragments
    const float*          __restrict__ bias,  // [256]
    float*                __restrict__ C,     // [MROWS, 256]
    int K) {
  const int ksteps = K >> 5;
  const int lane = threadIdx.x & 31;
  const int wave = threadIdx.x >> 5;
  const int gw   = blockIdx.x * 8 + wave;     // 10000 waves total
  const int tileM = gw >> 2;                  // 0..2499
  const int tng   = gw & 3;                   // N-tile group: cols [tng*64, +64)
  const int ln   = lane & 15;
  const int half = (lane >> 4) & 1;

  // A: lane holds row M = tileM*16 + ln; K offsets half*8 and half*8+16
  const unsigned short* arow = Abf + (size_t)(tileM * 16 + ln) * K + half * 8;
  // B: contiguous 8-dword fragment per lane per K-step; tile stride in dwords:
  const int tstride = ksteps * 256;           // ksteps*32 lanes*8 dwords
  const unsigned int* bbase =
      Wp + ((size_t)(tng * 4) * ksteps * 32 + lane) * 8;

  v8f acc0 = {}, acc1 = {}, acc2 = {}, acc3 = {};
  for (int kt = 0; kt < ksteps; ++kt) {
    const unsigned short* ap = arow + kt * 32;
    __builtin_prefetch(ap + 32, 0, 1);        // speculative: next K-tile of A
    v4u a0 = *(const v4u*)(ap);               // K = base .. base+7   (bf16 x8)
    v4u a1 = *(const v4u*)(ap + 16);          // K = base+16 .. base+23
    v8u av;
    av[0] = a0[0]; av[1] = a0[1]; av[2] = a0[2]; av[3] = a0[3];
    av[4] = a1[0]; av[5] = a1[1]; av[6] = a1[2]; av[7] = a1[3];
    v16bf afrag = __builtin_bit_cast(v16bf, av);

    const unsigned int* bk = bbase + (size_t)kt * 256;
    v8u bv0 = *(const v8u*)(bk);
    v8u bv1 = *(const v8u*)(bk + tstride);
    v8u bv2 = *(const v8u*)(bk + 2 * tstride);
    v8u bv3 = *(const v8u*)(bk + 3 * tstride);

    acc0 = __builtin_amdgcn_wmma_f32_16x16x32_bf16(
        false, afrag, false, __builtin_bit_cast(v16bf, bv0), (short)0, acc0,
        false, false);
    acc1 = __builtin_amdgcn_wmma_f32_16x16x32_bf16(
        false, afrag, false, __builtin_bit_cast(v16bf, bv1), (short)0, acc1,
        false, false);
    acc2 = __builtin_amdgcn_wmma_f32_16x16x32_bf16(
        false, afrag, false, __builtin_bit_cast(v16bf, bv2), (short)0, acc2,
        false, false);
    acc3 = __builtin_amdgcn_wmma_f32_16x16x32_bf16(
        false, afrag, false, __builtin_bit_cast(v16bf, bv3), (short)0, acc3,
        false, false);
  }

  // C/D layout: vgpr i -> row (half*8 + i), lane -> col
  const int rbase = tileM * 16 + half * 8;
  v8f accs[4] = {acc0, acc1, acc2, acc3};
#pragma unroll
  for (int t = 0; t < 4; ++t) {
    const int col = (tng * 4 + t) * 16 + ln;
    const float b = bias[col];
#pragma unroll
    for (int i = 0; i < 8; ++i) {
      float o = accs[t][i] + b;
      C[(size_t)(rbase + i) * HDIM + col] = o > 0.0f ? o : 0.0f;
    }
  }
}

// ---------------------------------------------------------------------------
// aggr[:, row[e], :] += h[:, col[e], :]  -- one block per edge, 256 channels
// ---------------------------------------------------------------------------
__global__ __launch_bounds__(256) void scatter_add_edges(
    const float* __restrict__ h, float* __restrict__ aggr,
    const int* __restrict__ ei, int E) {
  const int e = blockIdx.x;
  const int c = threadIdx.x;
  const int r = ei[e];        // edge_index[0, e]
  const int s = ei[E + e];    // edge_index[1, e]
#pragma unroll
  for (int b = 0; b < BATCH; ++b) {
    float v = h[((size_t)b * NNODE + s) * HDIM + c];
    __hip_atomic_fetch_add(&aggr[((size_t)b * NNODE + r) * HDIM + c], v,
                           __ATOMIC_RELAXED, __HIP_MEMORY_SCOPE_AGENT);
  }
}

// partial mean over nodes: 40 chunks of 250 nodes, atomic into g[b*256+c]
__global__ __launch_bounds__(256) void mean_partial_kernel(
    const float* __restrict__ h2, float* __restrict__ g) {
  const int c  = threadIdx.x;
  const int b  = blockIdx.y;
  const int n0 = blockIdx.x * 250;
  float s = 0.0f;
  for (int i = 0; i < 250; ++i)
    s += h2[((size_t)b * NNODE + n0 + i) * HDIM + c];
  __hip_atomic_fetch_add(&g[b * HDIM + c], s, __ATOMIC_RELAXED,
                         __HIP_MEMORY_SCOPE_AGENT);
}

// out[b] = relu((g[b]/N) @ Wc1 + bc1) @ Wc2 + bc2
__global__ __launch_bounds__(128) void classifier_kernel(
    const float* __restrict__ g,
    const float* __restrict__ Wc1, const float* __restrict__ bc1,
    const float* __restrict__ Wc2, const float* __restrict__ bc2,
    float* __restrict__ out) {
  const int b = blockIdx.x;
  const int j = threadIdx.x;                  // 0..127
  const float invN = 1.0f / (float)NNODE;
  float s = bc1[j];
  for (int k = 0; k < HDIM; ++k)
    s = fmaf(g[b * HDIM + k] * invN, Wc1[k * (HDIM / 2) + j], s);
  s = (s > 0.0f ? s : 0.0f) * Wc2[j];
  __shared__ float red[128];
  red[j] = s;
  __syncthreads();
  for (int st = 64; st > 0; st >>= 1) {
    if (j < st) red[j] += red[j + st];
    __syncthreads();
  }
  if (j == 0) out[b] = red[0] + bc2[0];
}

// ---------------------------------------------------------------------------
extern "C" void kernel_launch(void* const* d_in, const int* in_sizes, int n_in,
                              void* d_out, int out_size, void* d_ws,
                              size_t ws_size, hipStream_t stream) {
  const float* x       = (const float*)d_in[0];
  const int*   ei      = (const int*)d_in[1];
  const float* W_embed = (const float*)d_in[2];
  const float* b_embed = (const float*)d_in[3];
  const float* W1      = (const float*)d_in[4];
  const float* b1      = (const float*)d_in[5];
  const float* W2      = (const float*)d_in[6];
  const float* b2      = (const float*)d_in[7];
  const float* Wc1     = (const float*)d_in[8];
  const float* bc1     = (const float*)d_in[9];
  const float* Wc2     = (const float*)d_in[10];
  const float* bc2     = (const float*)d_in[11];
  float* out = (float*)d_out;
  const int E = in_sizes[1] / 2;

  // workspace layout (bytes)
  char* ws = (char*)d_ws;
  float*          bufA = (float*)(ws);                       // 40,960,000 B
  float*          bufB = (float*)(ws + 40960000);            // 40,960,000 B
  unsigned short* abf  = (unsigned short*)(ws + 81920000);   // 20,480,000 B
  unsigned int*   wp0  = (unsigned int*)(ws + 102400000);    //     65,536 B
  unsigned int*   wp1  = (unsigned int*)(ws + 102465536);    //    131,072 B
  unsigned int*   wp2  = (unsigned int*)(ws + 102596608);    //    131,072 B
  float*          g    = (float*)(ws + 102727680);           //      4,096 B

  // pack weights into WMMA B-fragment order (bf16)
  pack_w_kernel<<<64, 256, 0, stream>>>(W_embed, wp0, DIN);   // 16384 dwords
  pack_w_kernel<<<128, 256, 0, stream>>>(W1, wp1, HDIM);      // 32768 dwords
  pack_w_kernel<<<128, 256, 0, stream>>>(W2, wp2, HDIM);

  // layer 0: h = relu(x @ W_embed + b_embed)
  f32_to_bf16_kernel<<<(MROWS * DIN + 255) / 256, 256, 0, stream>>>(
      x, abf, MROWS * DIN);
  gemm_bias_relu_bf16<<<1250, 256, 0, stream>>>(abf, wp0, b_embed, bufA, DIN);

  // aggregate 1
  zero_f32_kernel<<<2048, 256, 0, stream>>>(bufB, (size_t)MROWS * HDIM);
  scatter_add_edges<<<E, 256, 0, stream>>>(bufA, bufB, ei, E);

  // layer 1: h1 = relu(aggr @ W1 + b1)
  f32_to_bf16_kernel<<<(MROWS * HDIM + 255) / 256, 256, 0, stream>>>(
      bufB, abf, MROWS * HDIM);
  gemm_bias_relu_bf16<<<1250, 256, 0, stream>>>(abf, wp1, b1, bufA, HDIM);

  // aggregate 2
  zero_f32_kernel<<<2048, 256, 0, stream>>>(bufB, (size_t)MROWS * HDIM);
  scatter_add_edges<<<E, 256, 0, stream>>>(bufA, bufB, ei, E);

  // layer 2: h2 = relu(aggr @ W2 + b2)
  f32_to_bf16_kernel<<<(MROWS * HDIM + 255) / 256, 256, 0, stream>>>(
      bufB, abf, MROWS * HDIM);
  gemm_bias_relu_bf16<<<1250, 256, 0, stream>>>(abf, wp2, b2, bufA, HDIM);

  // readout: mean over nodes, then tiny MLP classifier
  zero_f32_kernel<<<4, 256, 0, stream>>>(g, (size_t)BATCH * HDIM);
  mean_partial_kernel<<<dim3(40, BATCH), 256, 0, stream>>>(bufA, g);
  classifier_kernel<<<BATCH, 128, 0, stream>>>(g, Wc1, bc1, Wc2, bc2, out);
}
